// ObjectAttention_8727373546118
// MI455X (gfx1250) — matrix operand order repaired and checked by reading, not compiled, not run on hardware
//
#include <hip/hip_runtime.h>
#include <hip/hip_bf16.h>
#include <math.h>

#define BS   128
#define OBJ  512
#define RNN  1024
#define HID  512

typedef __attribute__((ext_vector_type(16))) __bf16 v16bf;
typedef __attribute__((ext_vector_type(8)))  float  v8f;

// ---------------------------------------------------------------------------
// Kernel 1: att_h[BS,HID] = h[BS,RNN] @ W[HID,RNN]^T + bias[HID]
// One wave per 16x16 output tile, v_wmma_f32_16x16x32_bf16, K-loop step 32.
// ---------------------------------------------------------------------------
__global__ __launch_bounds__(128) void h2att_gemm_wmma(
    const float* __restrict__ h, const float* __restrict__ W,
    const float* __restrict__ bias, float* __restrict__ att_h)
{
    const int lane  = threadIdx.x & 31;
    const int wave  = threadIdx.x >> 5;
    const int tile  = blockIdx.x * 4 + wave;   // 8 (M) x 32 (N) = 256 tiles
    const int tm    = (tile & 7)  * 16;        // M tile base
    const int tn    = (tile >> 3) * 16;        // N tile base
    const int l15   = lane & 15;
    const int hi    = lane >> 4;               // lane half: 0 or 1

    // A fragment source: row M = tm + l15; K offset 0/8 per lane half (ISA 7.12.2)
    const float* Arow = h + (size_t)(tm + l15) * RNN + hi * 8;
    // B fragment source: B[k][n] = W[n][k]; col N = tn + l15; K offset 0/8 per half
    const float* Brow = W + (size_t)(tn + l15) * RNN + hi * 8;

    // C/D layout: VGPR r -> M = tm + r + 8*hi, N = tn + l15. Bias depends on N only.
    v8f acc;
    const float bn = bias[tn + l15];
#pragma unroll
    for (int r = 0; r < 8; ++r) acc[r] = bn;

    for (int k0 = 0; k0 < RNN; k0 += 32) {
        // A per lane: elements 0..7 -> K = k0 + 8*hi + e ; 8..15 -> K = k0 + 16 + 8*hi + e
        float4 a0 = *(const float4*)(Arow + k0);
        float4 a1 = *(const float4*)(Arow + k0 + 4);
        float4 a2 = *(const float4*)(Arow + k0 + 16);
        float4 a3 = *(const float4*)(Arow + k0 + 20);
        // B per lane: elements 0..7 -> K = k0 + 8*hi + e ; 8..15 -> K = k0 + 16 + 8*hi + e
        float4 b0 = *(const float4*)(Brow + k0);
        float4 b1 = *(const float4*)(Brow + k0 + 4);
        float4 b2 = *(const float4*)(Brow + k0 + 16);
        float4 b3 = *(const float4*)(Brow + k0 + 20);

        float af[16] = {a0.x,a0.y,a0.z,a0.w, a1.x,a1.y,a1.z,a1.w,
                        a2.x,a2.y,a2.z,a2.w, a3.x,a3.y,a3.z,a3.w};
        float bf[16] = {b0.x,b0.y,b0.z,b0.w, b1.x,b1.y,b1.z,b1.w,
                        b2.x,b2.y,b2.z,b2.w, b3.x,b3.y,b3.z,b3.w};
        v16bf av, bv;
#pragma unroll
        for (int e = 0; e < 16; ++e) { av[e] = (__bf16)af[e]; bv[e] = (__bf16)bf[e]; }

        acc = __builtin_amdgcn_wmma_f32_16x16x32_bf16(
            /*neg_a=*/false, av, /*neg_b=*/false, bv,
            /*c_mod=*/(short)0, acc, /*reuse_a=*/false, /*reuse_b=*/false);
    }

#pragma unroll
    for (int r = 0; r < 8; ++r) {
        const int row = tm + r + hi * 8;
        att_h[(size_t)row * HID + tn + l15] = acc[r];
    }
}

// ---------------------------------------------------------------------------
// Kernel 2 (bandwidth-bound): scores[b,o] = sum_h tanh(feats[b,o,h]+att_h[b,h])*w[h] + b_alpha
// Block = 256 threads (8 waves); wave w handles object o = blockIdx.x*8 + w; b = blockIdx.y.
// att_h row + w_alpha staged in LDS; att_feats streamed as coalesced float4s.
// ---------------------------------------------------------------------------
__global__ __launch_bounds__(256) void score_kernel(
    const float* __restrict__ att_feats, const float* __restrict__ att_h,
    const float* __restrict__ w_alpha,  const float* __restrict__ b_alpha,
    float* __restrict__ scores)
{
    __shared__ __align__(16) float sh_ah[HID];
    __shared__ __align__(16) float sh_wa[HID];

    const int b = blockIdx.y;
    for (int i = threadIdx.x; i < HID; i += 256) {
        sh_ah[i] = att_h[(size_t)b * HID + i];
        sh_wa[i] = w_alpha[i];
    }
    __syncthreads();

    const int wave = threadIdx.x >> 5;
    const int lane = threadIdx.x & 31;
    const int o    = blockIdx.x * 8 + wave;

    const float* f = att_feats + ((size_t)b * OBJ + o) * HID;

    float acc = 0.f;
#pragma unroll
    for (int c = 0; c < 4; ++c) {
        const int idx = c * 128 + lane * 4;         // wave covers 128 contiguous floats
        float4 v = *(const float4*)(f + idx);
        float4 a = *(const float4*)(sh_ah + idx);
        float4 w = *(const float4*)(sh_wa + idx);
        acc += tanhf(v.x + a.x) * w.x;
        acc += tanhf(v.y + a.y) * w.y;
        acc += tanhf(v.z + a.z) * w.z;
        acc += tanhf(v.w + a.w) * w.w;
    }
    // wave32 reduction
#pragma unroll
    for (int off = 16; off > 0; off >>= 1)
        acc += __shfl_down(acc, off, 32);

    if (lane == 0)
        scores[(size_t)b * OBJ + o] = acc + b_alpha[0];
}

// ---------------------------------------------------------------------------
// Kernel 3: masked softmax + renorm. weight_i = m_i e^{s_i-max} / sum_j m_j e^{s_j-max}
// One block (512 threads) per batch row.
// ---------------------------------------------------------------------------
__global__ __launch_bounds__(512) void masked_softmax_kernel(
    const float* __restrict__ scores, const int* __restrict__ masks,
    float* __restrict__ out)
{
    __shared__ float red[OBJ];
    const int b = blockIdx.x;
    const int o = threadIdx.x;

    const float s = scores[(size_t)b * OBJ + o];
    red[o] = s;
    __syncthreads();
    for (int st = 256; st > 0; st >>= 1) {
        if (o < st) red[o] = fmaxf(red[o], red[o + st]);
        __syncthreads();
    }
    const float mx = red[0];
    __syncthreads();

    const float m  = (float)masks[(size_t)b * OBJ + o];
    const float em = expf(s - mx) * m;
    red[o] = em;
    __syncthreads();
    for (int st = 256; st > 0; st >>= 1) {
        if (o < st) red[o] += red[o + st];
        __syncthreads();
    }
    const float sum = red[0];

    out[(size_t)b * OBJ + o] = em / sum;
}

// ---------------------------------------------------------------------------
extern "C" void kernel_launch(void* const* d_in, const int* in_sizes, int n_in,
                              void* d_out, int out_size, void* d_ws, size_t ws_size,
                              hipStream_t stream)
{
    const float* h         = (const float*)d_in[0];   // [BS, RNN]
    const float* att_feats = (const float*)d_in[1];   // [BS, OBJ, HID]
    const int*   att_masks = (const int*)  d_in[2];   // [BS, OBJ]
    const float* W_h2att   = (const float*)d_in[3];   // [HID, RNN]
    const float* b_h2att   = (const float*)d_in[4];   // [HID]
    const float* w_alpha   = (const float*)d_in[5];   // [HID]
    const float* b_alpha   = (const float*)d_in[6];   // scalar

    float* att_h  = (float*)d_ws;                     // BS*HID floats (256 KB)
    float* scores = att_h + (size_t)BS * HID;         // BS*OBJ floats (256 KB)
    float* out    = (float*)d_out;                    // [BS, OBJ]

    // 1) att_h = h @ W^T + b  — 256 WMMA tiles, 4 waves/block -> 64 blocks
    h2att_gemm_wmma<<<64, 128, 0, stream>>>(h, W_h2att, b_h2att, att_h);

    // 2) scores — grid (OBJ/8, BS) = (64, 128), 8 waves/block
    dim3 sgrid(OBJ / 8, BS);
    score_kernel<<<sgrid, 256, 0, stream>>>(att_feats, att_h, w_alpha, b_alpha, scores);

    // 3) masked softmax — one block per batch row
    masked_softmax_kernel<<<BS, OBJ, 0, stream>>>(scores, att_masks, out);
}
